// GCNEncoder_76038101008750
// MI455X (gfx1250) — compile-verified, matrix operand order checked by reading, not run on hardware
//
#include <hip/hip_runtime.h>
#include <hip/hip_bf16.h>

// ---------------------------------------------------------------------------
// GCN encoder for MI455X (gfx1250, wave32).
//   per layer: h = act @ W.T            (WMMA bf16-split, f32 accumulate)
//              agg = bias; agg[dst] += h[src] * ew   (f32 global atomics, L2-resident)
//              act = relu(batchnorm(agg))            (fused elementwise)
// N=50000 nodes, C=256 channels -> h/agg are 51.2MB each: fit in 192MB L2,
// so the scatter phase runs at L2 bandwidth, not HBM.
// ---------------------------------------------------------------------------

typedef __attribute__((ext_vector_type(16))) __bf16 v16bf;
typedef __attribute__((ext_vector_type(8)))  __bf16 v8bf;
typedef __attribute__((ext_vector_type(4)))  __bf16 v4bf;
typedef __attribute__((ext_vector_type(8)))  float  v8f;

// 32 k-elements + 8 pad. 40 bf16 = 80 B = 5*16 B: keeps 16-byte alignment for
// b128 fragment loads, and row starts land on 20-bank multiples -> the 16 lane
// starts are distinct mod 64 banks and 4-bank disjoint (conflict-free).
#define LDS_STRIDE 40

// ---------------------------------------------------------------------------
// GEMM: Hout[N x 256] = A[N x K] * W[256 x K]^T, via split-bf16 WMMA.
// Block: 256 threads = 8 waves; wave w -> 16x16 tile (rowTile=w>>2, colTile=w&3)
// Block tile: 32 rows x 64 cols. grid = (ceil(N/32), 256/64).
// Both A and B tiles are stored k-contiguous per row/col in LDS so every
// fragment is two contiguous 16-byte runs -> ds_load_b128 pairs.
// ---------------------------------------------------------------------------
__global__ __launch_bounds__(256) void gcn_gemm_bf16x3(
    const float* __restrict__ A, const float* __restrict__ W,
    float* __restrict__ Hout, int Nrows, int K)
{
    __shared__ __bf16 Ahi[32 * LDS_STRIDE];
    __shared__ __bf16 Alo[32 * LDS_STRIDE];
    __shared__ __bf16 Bhi[64 * LDS_STRIDE];   // col-major: Bt[col][k]
    __shared__ __bf16 Blo[64 * LDS_STRIDE];

    const int tid  = threadIdx.x;
    const int wave = tid >> 5;
    const int lane = tid & 31;
    const int half = lane >> 4;      // K-half selector for 16-bit fragments
    const int lidx = lane & 15;      // row (A) / col (B,C,D) within tile

    const int rowBase = blockIdx.x * 32;
    const int colBase = blockIdx.y * 64;
    const int waveRow = (wave >> 2) * 16;   // 0 or 16
    const int waveCol = (wave & 3) * 16;    // 0..48

    v8f acc = {0.f, 0.f, 0.f, 0.f, 0.f, 0.f, 0.f, 0.f};

    // per-lane fragment base offsets (bf16 units); two 8-elem runs at +0, +16
    const int aoff = (waveRow + lidx) * LDS_STRIDE + half * 8;
    const int boff = (waveCol + lidx) * LDS_STRIDE + half * 8;

    for (int k = 0; k < K; k += 32) {
        // ---- stage A tile (32 rows x 32 k) with hi/lo bf16 split ----------
        {
            const int r  = tid >> 3;          // 0..31
            const int kk = (tid & 7) * 4;     // 0,4,...,28
            float4 v = {0.f, 0.f, 0.f, 0.f};
            const int gr = rowBase + r;
            if (gr < Nrows)
                v = *(const float4*)(A + (size_t)gr * K + k + kk);
            const float f[4] = {v.x, v.y, v.z, v.w};
            v4bf hi4, lo4;
            #pragma unroll
            for (int q = 0; q < 4; ++q) {
                __bf16 hi = (__bf16)f[q];
                hi4[q] = hi;
                lo4[q] = (__bf16)(f[q] - (float)hi);
            }
            *(v4bf*)(Ahi + r * LDS_STRIDE + kk) = hi4;   // 8B ds_store
            *(v4bf*)(Alo + r * LDS_STRIDE + kk) = lo4;
        }
        // ---- stage B tile col-major: Bt[col][kk] = W[colBase+col][k+kk] ---
        #pragma unroll
        for (int j = 0; j < 2; ++j) {
            const int idx = tid + j * 256;    // 0..511
            const int c   = idx >> 3;         // 0..63
            const int kk  = (idx & 7) * 4;
            float4 v = *(const float4*)(W + (size_t)(colBase + c) * K + k + kk);
            const float f[4] = {v.x, v.y, v.z, v.w};
            v4bf hi4, lo4;
            #pragma unroll
            for (int q = 0; q < 4; ++q) {
                __bf16 hi = (__bf16)f[q];
                hi4[q] = hi;
                lo4[q] = (__bf16)(f[q] - (float)hi);
            }
            *(v4bf*)(Bhi + c * LDS_STRIDE + kk) = hi4;
            *(v4bf*)(Blo + c * LDS_STRIDE + kk) = lo4;
        }
        __syncthreads();

        // ---- fragment loads: 2x 16-byte LDS loads each (ds_load_b128) -----
        // CDNA5 16-bit layout: VGPR v holds K pair {k0,k0+1},
        // k0 = (v<4 ? 2v : 16+2(v-4)) + 8*half -> elements 0..7 = K half*8+0..7,
        // elements 8..15 = K 16+half*8+0..7 : two contiguous runs.
        v8bf ah0 = *(const v8bf*)(Ahi + aoff);
        v8bf ah1 = *(const v8bf*)(Ahi + aoff + 16);
        v8bf al0 = *(const v8bf*)(Alo + aoff);
        v8bf al1 = *(const v8bf*)(Alo + aoff + 16);
        v8bf bh0 = *(const v8bf*)(Bhi + boff);
        v8bf bh1 = *(const v8bf*)(Bhi + boff + 16);
        v8bf bl0 = *(const v8bf*)(Blo + boff);
        v8bf bl1 = *(const v8bf*)(Blo + boff + 16);

        v16bf ah = __builtin_shufflevector(ah0, ah1, 0,1,2,3,4,5,6,7,8,9,10,11,12,13,14,15);
        v16bf al = __builtin_shufflevector(al0, al1, 0,1,2,3,4,5,6,7,8,9,10,11,12,13,14,15);
        v16bf bh = __builtin_shufflevector(bh0, bh1, 0,1,2,3,4,5,6,7,8,9,10,11,12,13,14,15);
        v16bf bl = __builtin_shufflevector(bl0, bl1, 0,1,2,3,4,5,6,7,8,9,10,11,12,13,14,15);

        // split-bf16 fp32 emulation: hi*hi + hi*lo + lo*hi, f32 accumulate
        acc = __builtin_amdgcn_wmma_f32_16x16x32_bf16(false, ah, false, bh,
                                                      (short)0, acc, false, false);
        acc = __builtin_amdgcn_wmma_f32_16x16x32_bf16(false, ah, false, bl,
                                                      (short)0, acc, false, false);
        acc = __builtin_amdgcn_wmma_f32_16x16x32_bf16(false, al, false, bh,
                                                      (short)0, acc, false, false);
        __syncthreads();
    }

    // ---- store D: VGPR r -> row (r + 8*half), lane&15 -> col --------------
    const int outCol = colBase + waveCol + lidx;
    #pragma unroll
    for (int r = 0; r < 8; ++r) {
        const int row = rowBase + waveRow + r + 8 * half;
        if (row < Nrows)
            Hout[(size_t)row * 256 + outCol] = acc[r];
    }
}

// ---------------------------------------------------------------------------
// ew[e] = mean(edge_attr[e, 0:8])
// ---------------------------------------------------------------------------
__global__ __launch_bounds__(256) void edge_mean_kernel(
    const float* __restrict__ attr, float* __restrict__ ew, int E)
{
    const int e = blockIdx.x * 256 + threadIdx.x;
    if (e >= E) return;
    float4 a = *(const float4*)(attr + (size_t)e * 8);
    float4 b = *(const float4*)(attr + (size_t)e * 8 + 4);
    ew[e] = (a.x + a.y + a.z + a.w + b.x + b.y + b.z + b.w) * 0.125f;
}

// ---------------------------------------------------------------------------
// agg[i, :] = bias  (float4-vectorized broadcast; re-inits scratch each call)
// ---------------------------------------------------------------------------
__global__ __launch_bounds__(256) void init_bias_kernel(
    float* __restrict__ out, const float* __restrict__ b, long long n4)
{
    const long long i = (long long)blockIdx.x * 256 + threadIdx.x;
    if (i >= n4) return;
    const int c = ((int)i & 63) * 4;       // channel group within 256
    float4 v = { b[c], b[c + 1], b[c + 2], b[c + 3] };
    *(float4*)(out + i * 4) = v;
}

// ---------------------------------------------------------------------------
// agg[dst[e], :] += h[src[e], :] * w_e   (64 lanes/edge, float4 gather,
// 4x global_atomic_add_f32 -> L2 atomic units; h & agg are L2-resident).
// global_prefetch of the gather row ~8 edges ahead hides gather latency.
// ---------------------------------------------------------------------------
__global__ __launch_bounds__(256) void scatter_add_kernel(
    const float* __restrict__ H, const int* __restrict__ src,
    const int* __restrict__ dst, const float* __restrict__ ew,
    float* __restrict__ agg, int E)
{
    const long long gid = (long long)blockIdx.x * 256 + threadIdx.x;
    const int e = (int)(gid >> 6);
    if (e >= E) return;
    const int q = ((int)gid & 63) * 4;
    const int s = src[e];
    const int d = dst[e];
    // prefetch the gather row of a future edge (same lane role, +8 edges)
    if (e + 8 < E) {
        const int sf = src[e + 8];
        __builtin_prefetch(H + (size_t)sf * 256 + q, 0, 0);
    }
    const float w = ew ? ew[e] : 1.0f;
    float4 v = *(const float4*)(H + (size_t)s * 256 + q);
    float* out = agg + (size_t)d * 256 + q;
    atomicAdd(out + 0, v.x * w);
    atomicAdd(out + 1, v.y * w);
    atomicAdd(out + 2, v.z * w);
    atomicAdd(out + 3, v.w * w);
}

// ---------------------------------------------------------------------------
// in-place x = relu((x - mean) * gamma * rsqrt(var + eps) + beta)
// ---------------------------------------------------------------------------
__global__ __launch_bounds__(256) void bn_relu_kernel(
    float* __restrict__ x, const float* __restrict__ g,
    const float* __restrict__ beta, const float* __restrict__ m,
    const float* __restrict__ var, long long n4)
{
    const long long i = (long long)blockIdx.x * 256 + threadIdx.x;
    if (i >= n4) return;
    const int c = ((int)i & 63) * 4;
    float4 v = *(float4*)(x + i * 4);
    float o[4] = {v.x, v.y, v.z, v.w};
    #pragma unroll
    for (int j = 0; j < 4; ++j) {
        const float inv = g[c + j] * rsqrtf(var[c + j] + 1e-5f);
        float r = (o[j] - m[c + j]) * inv + beta[c + j];
        o[j] = r > 0.f ? r : 0.f;
    }
    v.x = o[0]; v.y = o[1]; v.z = o[2]; v.w = o[3];
    *(float4*)(x + i * 4) = v;
}

// ---------------------------------------------------------------------------
extern "C" void kernel_launch(void* const* d_in, const int* in_sizes, int n_in,
                              void* d_out, int out_size, void* d_ws, size_t ws_size,
                              hipStream_t stream) {
    (void)n_in; (void)out_size; (void)ws_size;

    const float* x     = (const float*)d_in[0];
    const int*   ei    = (const int*)d_in[1];   // [2, E] int32 (JAX x64 off)
    const float* ea    = (const float*)d_in[2];
    const float* W1    = (const float*)d_in[3];
    const float* b1    = (const float*)d_in[4];
    const float* g1    = (const float*)d_in[5];
    const float* beta1 = (const float*)d_in[6];
    const float* m1    = (const float*)d_in[7];
    const float* v1    = (const float*)d_in[8];
    const float* W2    = (const float*)d_in[9];
    const float* b2    = (const float*)d_in[10];
    const float* g2    = (const float*)d_in[11];
    const float* beta2 = (const float*)d_in[12];
    const float* m2    = (const float*)d_in[13];
    const float* v2    = (const float*)d_in[14];
    const float* W3    = (const float*)d_in[15];
    const float* b3    = (const float*)d_in[16];

    const int Nn = in_sizes[0] / 128;     // 50000
    const int E  = in_sizes[1] / 2;       // 1600000
    const int*   src = ei;
    const int*   dst = ei + E;

    // workspace: ew[E] | bufA[N*256] | bufB[N*256]  (~109 MB)
    float* ews  = (float*)d_ws;
    float* bufA = ews + E;
    float* bufB = bufA + (size_t)Nn * 256;
    float* out  = (float*)d_out;

    const int rowBlocks = (Nn + 31) / 32;
    const dim3 gemmGrid(rowBlocks, 4);                  // 4 * 64 = 256 cols
    const long long n4 = (long long)Nn * 64;            // float4 groups
    const int elemBlocks = (int)((n4 + 255) / 256);
    const int scatBlocks = (int)(((long long)E * 64 + 255) / 256);
    const int edgeBlocks = (E + 255) / 256;

    // edge weights (shared by layers 1 & 2)
    edge_mean_kernel<<<edgeBlocks, 256, 0, stream>>>(ea, ews, E);

    // ---- layer 1: 128 -> 256 ----
    gcn_gemm_bf16x3<<<gemmGrid, 256, 0, stream>>>(x, W1, bufA, Nn, 128);
    init_bias_kernel<<<elemBlocks, 256, 0, stream>>>(bufB, b1, n4);
    scatter_add_kernel<<<scatBlocks, 256, 0, stream>>>(bufA, src, dst, ews, bufB, E);
    bn_relu_kernel<<<elemBlocks, 256, 0, stream>>>(bufB, g1, beta1, m1, v1, n4);

    // ---- layer 2: 256 -> 256 ----
    gcn_gemm_bf16x3<<<gemmGrid, 256, 0, stream>>>(bufB, W2, bufA, Nn, 256);
    init_bias_kernel<<<elemBlocks, 256, 0, stream>>>(bufB, b2, n4);
    scatter_add_kernel<<<scatBlocks, 256, 0, stream>>>(bufA, src, dst, ews, bufB, E);
    bn_relu_kernel<<<elemBlocks, 256, 0, stream>>>(bufB, g2, beta2, m2, v2, n4);

    // ---- layer 3: 256 -> 256, no edge weight, bias only ----
    gcn_gemm_bf16x3<<<gemmGrid, 256, 0, stream>>>(bufB, W3, bufA, Nn, 256);
    init_bias_kernel<<<elemBlocks, 256, 0, stream>>>(out, b3, n4);
    scatter_add_kernel<<<scatBlocks, 256, 0, stream>>>(bufA, src, dst, nullptr, out, E);
}
